// Transformer_89386859364427
// MI455X (gfx1250) — matrix operand order repaired and checked by reading, not compiled, not run on hardware
//
#include <hip/hip_runtime.h>
#include <hip/hip_bf16.h>

typedef __attribute__((ext_vector_type(16))) _Float16 v16h;
typedef __attribute__((ext_vector_type(8)))  float    v8f;

#define LAT    1024   // latent size D
#define SEQ    1024   // sequence length S
#define NTOK   2048   // B*S tokens
#define NNODES 255    // internal tree nodes
#define LW     16     // leaf width

union Frag {
  v16h h;
  uint4 u4[2];
  unsigned int u[8];
};

__device__ __forceinline__ float gelu_exact(float x) {
  return 0.5f * x * (1.f + erff(x * 0.70710678118654752f));
}

// ---------------------------------------------------------------------------
// Normalize: mu = mean, sd = std(ddof=1), xn = (x - mu) / sqrt(sd + 1e-5)
// one block (256 threads) per token
// ---------------------------------------------------------------------------
__global__ void norm_kernel(const float* __restrict__ x, float* __restrict__ xn) {
  __shared__ float r1[8], r2[8];
  __shared__ float smu, sinv;
  const int t = threadIdx.x;
  const float* xr = x + (size_t)blockIdx.x * LAT;
  float v[4]; float s = 0.f, s2 = 0.f;
#pragma unroll
  for (int i = 0; i < 4; i++) { float a = xr[t + 256 * i]; v[i] = a; s += a; s2 += a * a; }
#pragma unroll
  for (int o = 16; o > 0; o >>= 1) { s += __shfl_down(s, o, 32); s2 += __shfl_down(s2, o, 32); }
  if ((t & 31) == 0) { r1[t >> 5] = s; r2[t >> 5] = s2; }
  __syncthreads();
  if (t == 0) {
    float ss = 0.f, qq = 0.f;
    for (int w = 0; w < 8; w++) { ss += r1[w]; qq += r2[w]; }
    float mu  = ss * (1.f / LAT);
    float var = fmaxf((qq - (float)LAT * mu * mu) * (1.f / (LAT - 1)), 0.f);
    float sd  = sqrtf(var);
    smu  = mu;
    sinv = 1.f / sqrtf(sd + 1e-5f);
  }
  __syncthreads();
  const float mu = smu, inv = sinv;
  float* xo = xn + (size_t)blockIdx.x * LAT;
#pragma unroll
  for (int i = 0; i < 4; i++) xo[t + 256 * i] = (v[i] - mu) * inv;
}

// ---------------------------------------------------------------------------
// Hard-routed FFF (eval): 8 routing dot-products, then per-token leaf matvecs.
// MODE 0: act=ReLU, write f16 output (for q/k/v).
// MODE 1: act=exact GELU, write fp32 out = input + fff(input) (for final f).
// one block (256 threads, 8 waves) per token
// ---------------------------------------------------------------------------
template <int MODE>
__global__ void fff_kernel(const float* __restrict__ in,
                           const float* __restrict__ nw, const float* __restrict__ nb,
                           const float* __restrict__ w1, const float* __restrict__ b1,
                           const float* __restrict__ w2, const float* __restrict__ b2,
                           _Float16* __restrict__ outh, float* __restrict__ outf) {
  __shared__ float xs[LAT];
  __shared__ float red[8];
  __shared__ float hred[8][LW];
  __shared__ float hs[LW];
  __shared__ int   snode;
  const int t   = threadIdx.x;
  const int tok = blockIdx.x;
  const float* xr = in + (size_t)tok * LAT;
  for (int d = t; d < LAT; d += 256) xs[d] = xr[d];
  __syncthreads();

  // --- tree routing ---
  int node = 0;
  for (int step = 0; step < 8; ++step) {
    const float* nwp = nw + (size_t)node * LAT;
    float p = 0.f;
    for (int d = t; d < LAT; d += 256) p += xs[d] * nwp[d];
#pragma unroll
    for (int o = 16; o > 0; o >>= 1) p += __shfl_down(p, o, 32);
    if ((t & 31) == 0) red[t >> 5] = p;
    __syncthreads();
    if (t == 0) {
      float s = nb[node];
      for (int w = 0; w < 8; w++) s += red[w];
      snode = 2 * node + 1 + (s > 0.f ? 1 : 0);
    }
    __syncthreads();
    node = snode;
    __syncthreads();
  }
  const int leaf = node - NNODES;

  // --- h = act(x @ w1[leaf] + b1[leaf]) ---
  float acc[LW];
#pragma unroll
  for (int j = 0; j < LW; j++) acc[j] = 0.f;
  const float* w1l = w1 + (size_t)leaf * LAT * LW;
  for (int d = t; d < LAT; d += 256) {
    const float xv = xs[d];
    const float4* wr = (const float4*)(w1l + (size_t)d * LW);
#pragma unroll
    for (int q4 = 0; q4 < 4; q4++) {
      float4 w = wr[q4];
      acc[q4 * 4 + 0] += xv * w.x;
      acc[q4 * 4 + 1] += xv * w.y;
      acc[q4 * 4 + 2] += xv * w.z;
      acc[q4 * 4 + 3] += xv * w.w;
    }
  }
#pragma unroll
  for (int j = 0; j < LW; j++) {
    float p = acc[j];
#pragma unroll
    for (int o = 16; o > 0; o >>= 1) p += __shfl_down(p, o, 32);
    if ((t & 31) == 0) hred[t >> 5][j] = p;
  }
  __syncthreads();
  if (t < LW) {
    float s = b1[leaf * LW + t];
    for (int w = 0; w < 8; w++) s += hred[w][t];
    hs[t] = (MODE == 0) ? fmaxf(s, 0.f) : gelu_exact(s);
  }
  __syncthreads();

  // --- out = h @ w2[leaf] + b2[leaf] (+ residual for MODE 1) ---
  const float* w2l = w2 + (size_t)leaf * LW * LAT;
  const float* b2l = b2 + (size_t)leaf * LAT;
  for (int d = t; d < LAT; d += 256) {
    float v = b2l[d];
#pragma unroll
    for (int j = 0; j < LW; j++) v += hs[j] * w2l[j * LAT + d];
    if (MODE == 0) outh[(size_t)tok * LAT + d] = (_Float16)v;
    else           outf[(size_t)tok * LAT + d] = xs[d] + v;
  }
}

// ---------------------------------------------------------------------------
// scores = (Q @ K^T) / sqrt(D). One wave per 16x16 output tile; WMMA f16.
// A-frag: rows of Q; B-frag (=K^T): rows of K; both contiguous f16 loads.
// grid 1024 x 256 threads (8 waves/block, 8192 tiles total)
// ---------------------------------------------------------------------------
__global__ void scores_kernel(const _Float16* __restrict__ qh,
                              const _Float16* __restrict__ kh,
                              float* __restrict__ sc) {
  const int lane = threadIdx.x & 31;
  const int wid  = (blockIdx.x << 3) + (threadIdx.x >> 5);   // 0..8191
  const int b  = wid >> 12;
  const int qt = (wid >> 6) & 63;
  const int kt = wid & 63;
  const int g = lane >> 4;
  const int m = lane & 15;
  const _Float16* arow = qh + (size_t)(b * SEQ + qt * 16 + m) * LAT;
  const _Float16* brow = kh + (size_t)(b * SEQ + kt * 16 + m) * LAT;
  v8f acc = {};
  for (int dk = 0; dk < LAT; dk += 32) {
    Frag a, bb;
    a.u4[0]  = *(const uint4*)(arow + dk + g * 8);        // K = g*8 + 2i    (v0..3)
    a.u4[1]  = *(const uint4*)(arow + dk + 16 + g * 8);   // K = 16+g*8+2i  (v4..7)
    bb.u4[0] = *(const uint4*)(brow + dk + g * 8);        // K = g*8 + 2i    (v0..3)
    bb.u4[1] = *(const uint4*)(brow + dk + 16 + g * 8);   // K = 16+g*8+2i  (v4..7)
    acc = __builtin_amdgcn_wmma_f32_16x16x32_f16(false, a.h, false, bb.h,
                                                 (short)0, acc, false, false);
  }
  const float scale = 0.03125f;  // 1/sqrt(1024)
  float* orow = sc + ((size_t)b * SEQ + qt * 16) * SEQ + kt * 16 + m;
#pragma unroll
  for (int r = 0; r < 8; r++)
    orow[(size_t)(r + g * 8) * SEQ] = acc[r] * scale;     // M = r + 8g, N = m
}

// ---------------------------------------------------------------------------
// Row softmax over 1024 entries; one block per row. fp32 in -> f16 out (P).
// ---------------------------------------------------------------------------
__global__ void softmax_kernel(const float* __restrict__ sc,
                               _Float16* __restrict__ ph) {
  __shared__ float red[8];
  __shared__ float sbc;
  const int t = threadIdx.x;
  const float* row = sc + (size_t)blockIdx.x * SEQ;
  _Float16*    out = ph + (size_t)blockIdx.x * SEQ;
  float v[4];
  float m = -3.4e38f;
#pragma unroll
  for (int i = 0; i < 4; i++) { v[i] = row[t + 256 * i]; m = fmaxf(m, v[i]); }
#pragma unroll
  for (int o = 16; o > 0; o >>= 1) m = fmaxf(m, __shfl_down(m, o, 32));
  if ((t & 31) == 0) red[t >> 5] = m;
  __syncthreads();
  if (t == 0) { float mm = red[0]; for (int w = 1; w < 8; w++) mm = fmaxf(mm, red[w]); sbc = mm; }
  __syncthreads();
  m = sbc;
  __syncthreads();
  float s = 0.f;
#pragma unroll
  for (int i = 0; i < 4; i++) { v[i] = expf(v[i] - m); s += v[i]; }
#pragma unroll
  for (int o = 16; o > 0; o >>= 1) s += __shfl_down(s, o, 32);
  if ((t & 31) == 0) red[t >> 5] = s;
  __syncthreads();
  if (t == 0) { float ss = 0.f; for (int w = 0; w < 8; w++) ss += red[w]; sbc = 1.f / ss; }
  __syncthreads();
  const float inv = sbc;
#pragma unroll
  for (int i = 0; i < 4; i++) out[t + 256 * i] = (_Float16)(v[i] * inv);
}

// ---------------------------------------------------------------------------
// Transpose V (f16): vh (b, tok, d) -> vt (b, d, tok), 32x32 LDS tiles.
// grid = B * 32 * 32 blocks of 256 threads; fully coalesced both ways.
// ---------------------------------------------------------------------------
__global__ void vtrans_kernel(const _Float16* __restrict__ vh,
                              _Float16* __restrict__ vt) {
  __shared__ _Float16 tile[32][33];
  const int bid = blockIdx.x;
  const int b  = bid >> 10;
  const int r  = bid & 1023;
  const int ty = r >> 5;          // token-tile
  const int tx = r & 31;          // d-tile
  const int t  = threadIdx.x;
  const int lr = t >> 5;          // 0..7
  const int lc = t & 31;          // 0..31
#pragma unroll
  for (int i = 0; i < 4; i++) {
    int row = lr + i * 8;
    tile[row][lc] = vh[((size_t)b * SEQ + ty * 32 + row) * LAT + tx * 32 + lc];
  }
  __syncthreads();
#pragma unroll
  for (int i = 0; i < 4; i++) {
    int row = lr + i * 8;         // d within tile
    vt[((size_t)b * LAT + tx * 32 + row) * SEQ + ty * 32 + lc] = tile[lc][row];
  }
}

// ---------------------------------------------------------------------------
// y = xn + P @ V. One wave per 16x16 (q,d) tile; WMMA f16.
// A-frag: rows of P (f16); B-frag: rows of V^T; both contiguous b128 loads.
// ---------------------------------------------------------------------------
__global__ void pv_kernel(const _Float16* __restrict__ ph,
                          const _Float16* __restrict__ vt,
                          const float* __restrict__ xn,
                          float* __restrict__ y) {
  const int lane = threadIdx.x & 31;
  const int wid  = (blockIdx.x << 3) + (threadIdx.x >> 5);   // 0..8191
  const int b  = wid >> 12;
  const int qt = (wid >> 6) & 63;
  const int dt = wid & 63;
  const int g = lane >> 4;
  const int m = lane & 15;
  const _Float16* arow = ph + (size_t)(b * SEQ + qt * 16 + m) * SEQ;  // P row (q)
  const _Float16* brow = vt + ((size_t)b * LAT + dt * 16 + m) * SEQ;  // V^T row (d)
  v8f acc = {};
  for (int kk = 0; kk < SEQ; kk += 32) {
    Frag a, bb;
    a.u4[0]  = *(const uint4*)(arow + kk + g * 8);        // K = g*8 + 2i
    a.u4[1]  = *(const uint4*)(arow + kk + 16 + g * 8);   // K = 16+g*8+2i
    bb.u4[0] = *(const uint4*)(brow + kk + g * 8);        // K = g*8 + 2i
    bb.u4[1] = *(const uint4*)(brow + kk + 16 + g * 8);   // K = 16+g*8+2i
    acc = __builtin_amdgcn_wmma_f32_16x16x32_f16(false, a.h, false, bb.h,
                                                 (short)0, acc, false, false);
  }
  const size_t obase = ((size_t)b * SEQ + qt * 16) * LAT + dt * 16 + m;
#pragma unroll
  for (int r = 0; r < 8; r++) {
    size_t idx = obase + (size_t)(r + g * 8) * LAT;
    y[idx] = xn[idx] + acc[r];
  }
}

// ---------------------------------------------------------------------------
// Orchestration
// ---------------------------------------------------------------------------
extern "C" void kernel_launch(void* const* d_in, const int* in_sizes, int n_in,
                              void* d_out, int out_size, void* d_ws, size_t ws_size,
                              hipStream_t stream) {
  (void)in_sizes; (void)n_in; (void)out_size; (void)ws_size;
  const float* x = (const float*)d_in[0];
  const float* P[4][6];  // q,k,v,f x {nw, nb, w1, b1, w2, b2}
  for (int f = 0; f < 4; f++)
    for (int j = 0; j < 6; j++)
      P[f][j] = (const float*)d_in[1 + f * 6 + j];

  char* ws = (char*)d_ws;
  float*    xn = (float*)(ws);                                   //  8 MB @ 0
  float*    yb = (float*)(ws + (size_t)8  * 1024 * 1024);        //  8 MB @ 8
  _Float16* qh = (_Float16*)(ws + (size_t)16 * 1024 * 1024);     //  4 MB @ 16
  _Float16* kh = (_Float16*)(ws + (size_t)20 * 1024 * 1024);     //  4 MB @ 20
  _Float16* vh = (_Float16*)(ws + (size_t)24 * 1024 * 1024);     //  4 MB @ 24
  float*    sc = (float*)(ws + (size_t)28 * 1024 * 1024);        //  8 MB @ 28
  _Float16* ph = (_Float16*)(ws + (size_t)36 * 1024 * 1024);     //  4 MB @ 36
  _Float16* vt = (_Float16*)(ws + (size_t)40 * 1024 * 1024);     //  4 MB @ 40

  norm_kernel<<<NTOK, 256, 0, stream>>>(x, xn);
  fff_kernel<0><<<NTOK, 256, 0, stream>>>(xn, P[0][0], P[0][1], P[0][2], P[0][3], P[0][4], P[0][5], qh, nullptr);
  fff_kernel<0><<<NTOK, 256, 0, stream>>>(xn, P[1][0], P[1][1], P[1][2], P[1][3], P[1][4], P[1][5], kh, nullptr);
  fff_kernel<0><<<NTOK, 256, 0, stream>>>(xn, P[2][0], P[2][1], P[2][2], P[2][3], P[2][4], P[2][5], vh, nullptr);
  scores_kernel<<<1024, 256, 0, stream>>>(qh, kh, sc);
  softmax_kernel<<<NTOK, 256, 0, stream>>>(sc, ph);
  vtrans_kernel<<<NTOK, 256, 0, stream>>>(vh, vt);
  pv_kernel<<<1024, 256, 0, stream>>>(ph, vt, xn, yb);
  fff_kernel<1><<<NTOK, 256, 0, stream>>>(yb, P[3][0], P[3][1], P[3][2], P[3][3], P[3][4], P[3][5], nullptr, (float*)d_out);
}
